// LSHAttention_66099546685776
// MI455X (gfx1250) — matrix operand order repaired and checked by reading, not compiled, not run on hardware
//
#include <hip/hip_runtime.h>
#include <hip/hip_bf16.h>
#include <math.h>

// ---------------------------------------------------------------------------
// Problem constants (from the reference)
// ---------------------------------------------------------------------------
#define BB     4
#define TT     4096
#define CC     2048
#define HH     16
#define FF     128
#define BH     (BB * HH)          // 64
#define HF     (HH * FF)          // 2048
#define NROW   (BB * TT)          // 16384
#define ROUNDS 6                  // (T-1).bit_length() - 6

typedef __attribute__((ext_vector_type(16))) __bf16        v16bf;
typedef __attribute__((ext_vector_type(8)))  float         v8f;
typedef __attribute__((ext_vector_type(8)))  unsigned int  v8u;

#define WMMA_BF16(a, b, c) \
    __builtin_amdgcn_wmma_f32_16x16x32_bf16(false, (a), false, (b), (short)0, (c), false, false)

// ---------------------------------------------------------------------------
// Helpers
// ---------------------------------------------------------------------------
__device__ __forceinline__ unsigned short f2bf(float f) {
    unsigned int u = __builtin_bit_cast(unsigned int, f);
    unsigned int r = u + 0x7fffu + ((u >> 16) & 1u);   // round-to-nearest-even
    if ((u & 0x7f800000u) == 0x7f800000u) r = u;       // keep inf/NaN
    return (unsigned short)(r >> 16);
}

// A-matrix fragment: 16x32 bf16 tile, source row-major (stride ld elements),
// p points at element (row0, k0).  Layout: lane<16 -> M=lane, K 0..15;
// lane>=16 -> M=lane-16, K 16..31.
__device__ __forceinline__ v16bf load_frag_A(const unsigned short* p, int ld, int lane) {
    v8u u;
    int m    = lane & 15;
    int koff = (lane >> 4) << 4;             // 0 or 16
    const unsigned short* rp = p + m * ld;
#pragma unroll
    for (int v = 0; v < 8; ++v) {
        int kk = koff + (v << 1);
        u[v] = *(const unsigned int*)(rp + kk);        // bf16 pair (kk, kk+1)
    }
    return __builtin_bit_cast(v16bf, u);
}

// B-matrix fragment: 32x16 bf16 where B(k,n) = src[n*ld + k]  (source holds
// the 16 output columns as rows, contiguous along k).
__device__ __forceinline__ v16bf load_frag_B_t(const unsigned short* p, int ld, int lane) {
    v8u u;
    int n  = lane & 15;
    int kb = (lane >> 4) << 4;               // 0 or 16
    const unsigned short* rp = p + n * ld + kb;
#pragma unroll
    for (int v = 0; v < 8; ++v)
        u[v] = *(const unsigned int*)(rp + (v << 1));  // K pair (kb+2v, kb+2v+1)
    return __builtin_bit_cast(v16bf, u);
}

// B-matrix fragment: 32x16 bf16 where B(k,n) = src[k*ld + n] (natural K-major,
// k not contiguous -> pack two 16-bit loads).
__device__ __forceinline__ v16bf load_frag_B_n(const unsigned short* p, int ld, int lane) {
    v8u u;
    int n  = lane & 15;
    int kb = (lane >> 4) << 4;
    const unsigned short* cp = p + kb * ld + n;
#pragma unroll
    for (int v = 0; v < 8; ++v) {
        unsigned int lo = cp[(2 * v) * ld];
        unsigned int hi = cp[(2 * v + 1) * ld];
        u[v] = lo | (hi << 16);
    }
    return __builtin_bit_cast(v16bf, u);
}

// ---------------------------------------------------------------------------
// 1) fp32 -> bf16 conversion
// ---------------------------------------------------------------------------
__global__ void cvt_bf16_kernel(const float* __restrict__ in,
                                unsigned short* __restrict__ out, int n) {
    int i = blockIdx.x * blockDim.x + threadIdx.x;
    if (i < n) out[i] = f2bf(in[i]);
}

// ---------------------------------------------------------------------------
// 2) Projection GEMM: Out(b,h,t,f) = sum_c x[b*T+t, c] * W[h*F+f, c] + bias
//    bf16 WMMA, f32 accumulate.  256 threads = 8 waves.  Each wave register-
//    blocks a 32x64 output tile (2 A frags x 4 B frags -> 8 WMMA per k-step,
//    0.75 KB fragment traffic per WMMA).  Block covers 128x128.
// ---------------------------------------------------------------------------
__global__ void __launch_bounds__(256)
gemm_bf16_kernel(const unsigned short* __restrict__ X,
                 const unsigned short* __restrict__ W,
                 const float* __restrict__ bias,
                 float* __restrict__ Out) {
    int lane = threadIdx.x & 31;
    int wave = threadIdx.x >> 5;
    int row0 = blockIdx.y * 128 + (wave & 3) * 32;     // 4 row groups of 32
    int col0 = blockIdx.x * 128 + (wave >> 2) * 64;    // 2 col groups of 64

    const unsigned short* X0 = X + (size_t)row0 * CC;
    const unsigned short* X1 = X + (size_t)(row0 + 16) * CC;
    const unsigned short* W0 = W + (size_t)col0 * CC;

    v8f acc[8] = {};                                   // acc[rr*4 + cc]
    for (int k0 = 0; k0 < CC; k0 += 32) {
        __builtin_prefetch(X0 + k0 + 512, 0, 1);       // global_prefetch_b8
        v16bf a0 = load_frag_A(X0 + k0, CC, lane);
        v16bf a1 = load_frag_A(X1 + k0, CC, lane);
        v16bf b0 = load_frag_B_t(W0 + k0, CC, lane);
        v16bf b1 = load_frag_B_t(W0 + 16 * CC + k0, CC, lane);
        v16bf b2 = load_frag_B_t(W0 + 32 * CC + k0, CC, lane);
        v16bf b3 = load_frag_B_t(W0 + 48 * CC + k0, CC, lane);
        acc[0] = WMMA_BF16(a0, b0, acc[0]);
        acc[1] = WMMA_BF16(a0, b1, acc[1]);
        acc[2] = WMMA_BF16(a0, b2, acc[2]);
        acc[3] = WMMA_BF16(a0, b3, acc[3]);
        acc[4] = WMMA_BF16(a1, b0, acc[4]);
        acc[5] = WMMA_BF16(a1, b1, acc[5]);
        acc[6] = WMMA_BF16(a1, b2, acc[6]);
        acc[7] = WMMA_BF16(a1, b3, acc[7]);
    }
    int nl = lane & 15;
#pragma unroll
    for (int cc = 0; cc < 4; ++cc) {
        int n = col0 + cc * 16 + nl;
        float bvv = bias[n];
        int h = n >> 7, f = n & 127;                   // n = h*F + f
#pragma unroll
        for (int rr = 0; rr < 2; ++rr) {
            v8f a = acc[rr * 4 + cc];
#pragma unroll
            for (int v = 0; v < 8; ++v) {
                int m = row0 + rr * 16 + v + ((lane >> 4) << 3);   // m = b*T + t
                int b = m >> 12, t = m & (TT - 1);
                Out[(((b << 4) + h) * TT + t) * FF + f] = a[v] + bvv;
            }
        }
    }
}

// ---------------------------------------------------------------------------
// 3) RoPE on K, in place.  K layout: (bh, t, F)
// ---------------------------------------------------------------------------
__global__ void rope_kernel(float* __restrict__ K, const float* __restrict__ alpha) {
    int i  = blockIdx.x * blockDim.x + threadIdx.x;    // over bh*T*64
    int j  = i & 63;
    int t  = (i >> 6) & (TT - 1);
    int bh = i >> 18;
    int h  = bh & (HH - 1);
    float* row = K + ((long)bh * TT + t) * FF;
    float kr = row[j], ki = row[64 + j];
    float P = (float)t * exp2f(-(float)j) * alpha[h * 64 + j];
    float s, c;
    sincosf(P, &s, &c);
    row[j]      = kr * c - ki * s;
    row[64 + j] = kr * s + ki * c;
}

// ---------------------------------------------------------------------------
// 4) LSH projections: proj[r][bh][t] = dot(K[bh,t,:], panels[r,bh,:])
// ---------------------------------------------------------------------------
__global__ void __launch_bounds__(256)
proj_kernel(const float* __restrict__ K, const float* __restrict__ panels,
            float* __restrict__ proj) {
    __shared__ float pan[FF];
    int round = blockIdx.z, bh = blockIdx.y;
    if (threadIdx.x < FF)
        pan[threadIdx.x] = panels[((long)round * BH + bh) * FF + threadIdx.x];
    __syncthreads();
    int t = blockIdx.x * blockDim.x + threadIdx.x;
    const float* row = K + ((long)bh * TT + t) * FF;
    float s = 0.f;
#pragma unroll 8
    for (int f = 0; f < FF; ++f) s += row[f] * pan[f];
    proj[((long)round * BH + bh) * TT + t] = s;
}

// ---------------------------------------------------------------------------
// 5) Segmented bitonic index sort.  Round r sorts each contiguous segment of
//    length T>>r ascending by key = proj[r][bh][idx], tie-broken by idx.
// ---------------------------------------------------------------------------
__global__ void __launch_bounds__(1024)
lsh_sort_kernel(const float* __restrict__ proj, int* __restrict__ perm) {
    __shared__ float key[TT];
    __shared__ int   idx[TT];
    int bh = blockIdx.x;
    for (int i = threadIdx.x; i < TT; i += 1024) idx[i] = i;
    for (int round = 0; round < ROUNDS; ++round) {
        __syncthreads();
        const float* pr = proj + ((long)round * BH + bh) * TT;
        for (int i = threadIdx.x; i < TT; i += 1024) key[i] = pr[idx[i]];
        __syncthreads();
        int seg = TT >> round;
        for (int k = 2; k <= seg; k <<= 1) {
            for (int j = k >> 1; j > 0; j >>= 1) {
                for (int i = threadIdx.x; i < TT; i += 1024) {
                    int ixj = i ^ j;
                    if (ixj > i) {
                        bool up = (((i & (seg - 1)) & k) == 0);   // per-segment ascending
                        float ka = key[i], kb = key[ixj];
                        int   ia = idx[i], ib = idx[ixj];
                        bool  gt = (ka > kb) || (ka == kb && ia > ib);
                        if (gt == up) {
                            key[i] = kb; key[ixj] = ka;
                            idx[i] = ib; idx[ixj] = ia;
                        }
                    }
                }
                __syncthreads();
            }
        }
    }
    for (int i = threadIdx.x; i < TT; i += 1024) perm[(long)bh * TT + i] = idx[i];
}

// ---------------------------------------------------------------------------
// 6) Bucketed attention.  One block per bucket (BH*64 = 4096 blocks).
//    S = Kb.Kb^T/sqrt(F) with diag+column masking, softmax, O = A.Vb,
//    fixer rows pass V through; fully-masked buckets output 0.
//    NOTE: reference writes output in *bucket order* (no inverse perm).
//    Waves are assigned tiles sharing one A fragment so the A load amortizes
//    over 2 (S stage) / 4 (O stage) WMMAs.
// ---------------------------------------------------------------------------
__global__ void __launch_bounds__(256)
lsh_attn_kernel(const float* __restrict__ Kb, const float* __restrict__ Vb,
                const unsigned char* __restrict__ mask,
                const int* __restrict__ perm, float* __restrict__ out) {
    __shared__ int            pidx[64];
    __shared__ unsigned char  mcol[64];
    __shared__ int            sUnmask;
    __shared__ __align__(16) unsigned short Kbf[64 * FF];   // 16 KB
    __shared__ __align__(16) unsigned short Vbf[64 * FF];   // 16 KB
    __shared__ __align__(16) float          S[64 * 64];     // 16 KB
    __shared__ __align__(16) unsigned short Abf[64 * 64];   // 8 KB

    int blk = blockIdx.x;
    int bh = blk >> 6, bucket = blk & 63;
    int b = bh >> 4, h = bh & 15;
    int tid = threadIdx.x, lane = tid & 31, wave = tid >> 5;

    if (tid < 64) {
        int p = perm[(long)bh * TT + bucket * 64 + tid];
        pidx[tid] = p;
        mcol[tid] = mask[b * TT + p];
    }
    __syncthreads();

    for (int e = tid; e < 64 * FF; e += 256) {
        int r = e >> 7, f = e & 127;
        long src = ((long)bh * TT + pidx[r]) * FF + f;
        Kbf[e] = f2bf(Kb[src]);
        Vbf[e] = f2bf(Vb[src]);
    }
    if (tid == 0) {
        int c = 0;
        for (int i = 0; i < 64; ++i) c += mcol[i] ? 0 : 1;
        sUnmask = c;
    }
    __syncthreads();

    // ---- S = K.K^T with mask/fixer/scale applied on the way to LDS ----
    // wave w: row tile tr = w>>1, col tiles tc = (w&1)*2 .. +1 (A frag shared)
    const float invScale = 0.08838834764831843f;            // 1/sqrt(128)
    const float NEGINF = -__builtin_inff();
    {
        int tr  = wave >> 1;
        int tc0 = (wave & 1) * 2;
        v8f acc0 = {}, acc1 = {};
#pragma unroll
        for (int kc = 0; kc < FF; kc += 32) {
            v16bf a  = load_frag_A(Kbf + tr * 16 * FF + kc, FF, lane);
            v16bf b0 = load_frag_B_t(Kbf + (tc0 + 0) * 16 * FF + kc, FF, lane);
            v16bf b1 = load_frag_B_t(Kbf + (tc0 + 1) * 16 * FF + kc, FF, lane);
            acc0 = WMMA_BF16(a, b0, acc0);
            acc1 = WMMA_BF16(a, b1, acc1);
        }
        int un = sUnmask;
#pragma unroll
        for (int q = 0; q < 2; ++q) {
            v8f acc = q ? acc1 : acc0;
            int c = (tc0 + q) * 16 + (lane & 15);
#pragma unroll
            for (int v = 0; v < 8; ++v) {
                int r = tr * 16 + v + ((lane >> 4) << 3);
                bool fixer = (un - (mcol[r] ? 0 : 1)) == 0;
                float sc;
                if (fixer)                    sc = 0.0f;
                else if (r == c || mcol[c])   sc = NEGINF;
                else                          sc = acc[v] * invScale;
                S[r * 64 + c] = sc;
            }
        }
    }
    __syncthreads();

    // ---- row softmax (one thread per row) ----
    if (tid < 64) {
        int r = tid;
        float mx = NEGINF;
        for (int c = 0; c < 64; ++c) mx = fmaxf(mx, S[r * 64 + c]);
        float sum = 0.f;
        for (int c = 0; c < 64; ++c) {
            float e = expf(S[r * 64 + c] - mx);
            S[r * 64 + c] = e;
            sum += e;
        }
        float inv = 1.0f / sum;
        for (int c = 0; c < 64; ++c) Abf[r * 64 + c] = f2bf(S[r * 64 + c] * inv);
    }
    __syncthreads();

    // ---- O = A.V, write in bucket order ----
    // wave w: row tile tr = w>>1, col tiles tcf = (w&1)*4 .. +3 (A shared x4)
    {
        int tr   = wave >> 1;
        int tcf0 = (wave & 1) * 4;
        v8f acc[4] = {};
#pragma unroll
        for (int kc = 0; kc < 64; kc += 32) {
            v16bf a = load_frag_A(Abf + tr * 16 * 64 + kc, 64, lane);
#pragma unroll
            for (int q = 0; q < 4; ++q) {
                v16bf bfr = load_frag_B_n(Vbf + kc * FF + (tcf0 + q) * 16, FF, lane);
                acc[q] = WMMA_BF16(a, bfr, acc[q]);
            }
        }
        int un = sUnmask;
        bool allMasked = (un == 0);
#pragma unroll
        for (int q = 0; q < 4; ++q) {
            int fcol = (tcf0 + q) * 16 + (lane & 15);
#pragma unroll
            for (int v = 0; v < 8; ++v) {
                int r = tr * 16 + v + ((lane >> 4) << 3);
                bool fixer = (un - (mcol[r] ? 0 : 1)) == 0;
                float val = acc[q][v];
                if (fixer) val = Vb[((long)bh * TT + pidx[r]) * FF + fcol];
                if (allMasked) val = 0.0f;
                int pos = bucket * 64 + r;           // reference keeps bucket order
                out[((long)b * TT + pos) * HF + h * FF + fcol] = val;
            }
        }
    }
}

// ---------------------------------------------------------------------------
// Host-side launcher
// ---------------------------------------------------------------------------
extern "C" void kernel_launch(void* const* d_in, const int* in_sizes, int n_in,
                              void* d_out, int out_size, void* d_ws, size_t ws_size,
                              hipStream_t stream) {
    const float*         x     = (const float*)d_in[0];
    const unsigned char* mask  = (const unsigned char*)d_in[1];
    const float*         Wk    = (const float*)d_in[2];
    const float*         bk    = (const float*)d_in[3];
    const float*         Wv    = (const float*)d_in[4];
    const float*         bv    = (const float*)d_in[5];
    const float*         alpha = (const float*)d_in[6];
    const float*         panels= (const float*)d_in[7];
    float*               out   = (float*)d_out;

    char* w = (char*)d_ws;
    auto carve = [&](size_t bytes) {
        char* p = w;
        w += (bytes + 255) & ~(size_t)255;
        return p;
    };
    unsigned short* xbf  = (unsigned short*)carve((size_t)NROW * CC * 2);
    unsigned short* wkbf = (unsigned short*)carve((size_t)HF * CC * 2);
    unsigned short* wvbf = (unsigned short*)carve((size_t)HF * CC * 2);
    float*          Kbuf = (float*)carve((size_t)BH * TT * FF * 4);
    float*          Vbuf = (float*)carve((size_t)BH * TT * FF * 4);
    float*          proj = (float*)carve((size_t)ROUNDS * BH * TT * 4);
    int*            perm = (int*)carve((size_t)BH * TT * 4);

    // 1) downconvert to bf16
    {
        int n = NROW * CC;
        cvt_bf16_kernel<<<(n + 255) / 256, 256, 0, stream>>>(x, xbf, n);
        n = HF * CC;
        cvt_bf16_kernel<<<(n + 255) / 256, 256, 0, stream>>>(Wk, wkbf, n);
        cvt_bf16_kernel<<<(n + 255) / 256, 256, 0, stream>>>(Wv, wvbf, n);
    }
    // 2) projection GEMMs (WMMA bf16, 128x128 blocks, 32x64 per wave)
    {
        dim3 grid(HF / 128, NROW / 128);
        gemm_bf16_kernel<<<grid, 256, 0, stream>>>(xbf, wkbf, bk, Kbuf);
        gemm_bf16_kernel<<<grid, 256, 0, stream>>>(xbf, wvbf, bv, Vbuf);
    }
    // 3) RoPE on K
    {
        int n = BH * TT * 64;
        rope_kernel<<<n / 256, 256, 0, stream>>>(Kbuf, alpha);
    }
    // 4) LSH projections
    {
        dim3 grid(TT / 256, BH, ROUNDS);
        proj_kernel<<<grid, 256, 0, stream>>>(Kbuf, panels, proj);
    }
    // 5) segmented bitonic index sort
    lsh_sort_kernel<<<BH, 1024, 0, stream>>>(proj, perm);
    // 6) bucketed attention (WMMA bf16) + bucket-order writeout
    lsh_attn_kernel<<<BH * 64, 256, 0, stream>>>(Kbuf, Vbuf, mask, perm, out);
}
